// SAFM9_41266045780173
// MI455X (gfx1250) — matrix operand order repaired and checked by reading, not compiled, not run on hardware
//
#include <hip/hip_runtime.h>
#include <hip/hip_bf16.h>
#include <math.h>

// ---------------------------------------------------------------------------
// Shapes: x [4,512,128,128] f32.  cat = [x1(256ch full-res dwconv),
// x2(128ch), x3(64ch), x4(64ch)] where x2..x4 are constant per 64x64 quadrant.
// agg = wa(512x512) @ cat per pixel, + ba, gelu(exact), * x.
// GEMM path: bf16 WMMA 16x16x32, fp32 accumulate.
// ---------------------------------------------------------------------------

typedef __attribute__((ext_vector_type(8)))  __bf16 v8bf;
typedef __attribute__((ext_vector_type(16))) __bf16 v16bf;
typedef __attribute__((ext_vector_type(8)))  float  v8f;

#define HW   16384   // 128*128
#define LDSK 520     // 512 + 8 pad (staggers LDS banks, keeps 16B alignment)

__device__ __forceinline__ unsigned short f2bf(float f) {
    unsigned u = __float_as_uint(f);
    u += 0x7FFFu + ((u >> 16) & 1u);          // round-to-nearest-even
    return (unsigned short)(u >> 16);
}

__device__ __forceinline__ v16bf ldpair(const unsigned short* p0,
                                        const unsigned short* p1) {
    v8bf lo = *(const v8bf*)p0;
    v8bf hi = *(const v8bf*)p1;
    return __builtin_shufflevector(lo, hi, 0,1,2,3,4,5,6,7,8,9,10,11,12,13,14,15);
}

// --------------------------- pass 0: wa -> bf16 ----------------------------
__global__ void safm_cvt_wa(const float* __restrict__ wa,
                            unsigned short* __restrict__ wa_bf) {
    int i = blockIdx.x * 256 + threadIdx.x;   // 512*512 elements, grid 1024
    wa_bf[i] = f2bf(wa[i]);
}

// ------------------- pass 1: quadrant max of channels 256..511 -------------
// grid 4096 = b(4) * c(256) * q(4); p2 layout ((b*256+c)*2+qh)*2+qw
__global__ void safm_pool_max(const float* __restrict__ x,
                              float* __restrict__ p2) {
    int bid = blockIdx.x, tid = threadIdx.x;
    int q = bid & 3, c = (bid >> 2) & 255, b = bid >> 10;
    int qw = q & 1, qh = q >> 1;
    const float* base = x + (((size_t)b * 512 + 256 + c) * 128 + qh * 64) * 128 + qw * 64;
    float m = -3.402823466e38f;
    for (int i = tid; i < 4096; i += 256) {
        int r = i >> 6, col = i & 63;
        m = fmaxf(m, base[r * 128 + col]);
    }
    __shared__ float s[256];
    s[tid] = m;
    __syncthreads();
    for (int st = 128; st > 0; st >>= 1) {
        if (tid < st) s[tid] = fmaxf(s[tid], s[tid + st]);
        __syncthreads();
    }
    if (tid == 0) p2[bid] = s[0];
}

// -------- pass 2: 3x3 dwconv on 2x2 grids (levels 2..4) -> coarse ----------
// coarse[((b*256+cc)*2+i)*2+j] == cat channel (256+cc) at quadrant (i,j)
__global__ void safm_coarse(const float* __restrict__ p2,
                            const float* __restrict__ w2, const float* __restrict__ b2,
                            const float* __restrict__ w3, const float* __restrict__ b3,
                            const float* __restrict__ w4, const float* __restrict__ b4,
                            float* __restrict__ coarse) {
    int cc = threadIdx.x;                      // 0..255
    const float* wp; float bias;
    if (cc < 128)      { wp = w2 + cc * 9;         bias = b2[cc];       }
    else if (cc < 192) { wp = w3 + (cc - 128) * 9; bias = b3[cc - 128]; }
    else               { wp = w4 + (cc - 192) * 9; bias = b4[cc - 192]; }
    for (int b = 0; b < 4; ++b) {
        const float* pp = p2 + ((size_t)b * 256 + cc) * 4;
        float in[2][2] = {{pp[0], pp[1]}, {pp[2], pp[3]}};
        for (int i = 0; i < 2; ++i)
            for (int j = 0; j < 2; ++j) {
                float a = bias;
                #pragma unroll
                for (int di = -1; di <= 1; ++di) {
                    int ii = i + di; if (ii < 0 || ii > 1) continue;
                    #pragma unroll
                    for (int dj = -1; dj <= 1; ++dj) {
                        int jj = j + dj; if (jj < 0 || jj > 1) continue;
                        a += wp[(di + 1) * 3 + (dj + 1)] * in[ii][jj];
                    }
                }
                coarse[((size_t)b * 256 + cc) * 4 + i * 2 + j] = a;
            }
    }
}

// ----------------- pass 3: fused cat-build + WMMA GEMM + epilogue ----------
// grid 1024 = b(4) * h(128) * wt(2); block 256 = 8 waves.
// Tile: all 512 out channels x 64 pixels (one row segment). wave w owns
// M rows [64w, 64w+64) -> 4x4 WMMA accumulators; K = 512 in 16 steps of 32.
__global__ void __launch_bounds__(256)
safm_main(const float* __restrict__ x,
          const float* __restrict__ w1, const float* __restrict__ b1,
          const float* __restrict__ coarse,
          const unsigned short* __restrict__ wa_bf,
          const float* __restrict__ ba,
          float* __restrict__ out) {
    __shared__ unsigned short cat_s[64 * LDSK];   // [pixel n][channel k] bf16

    const int tid = threadIdx.x;
    const int bid = blockIdx.x;
    const int wt = bid & 1;              // which 64-wide half of the row
    const int h  = (bid >> 1) & 127;
    const int b  = bid >> 8;
    const int w0 = wt * 64;

    // ---- build cat channels 0..255: depthwise 3x3 at full res -------------
    {
        const int wl = tid & 63;                 // pixel within tile
        const int csub = tid >> 2 ? 0 : 0;       // (placeholder, see below)
        (void)csub;
        const int cs = tid >> 6;                 // 0..3
        const int w = w0 + wl;
        const float* xb = x + (size_t)b * 512 * HW;
        for (int cb = 0; cb < 64; ++cb) {
            const int c = cb * 4 + cs;           // uniform within a wave
            const float* xc = xb + (size_t)c * HW;
            const float* wp = w1 + c * 9;
            float acc = b1[c];
            #pragma unroll
            for (int dh = -1; dh <= 1; ++dh) {
                const int hh = h + dh;
                if (hh < 0 || hh > 127) continue;
                #pragma unroll
                for (int dw = -1; dw <= 1; ++dw) {
                    const int ww = w + dw;
                    if (ww < 0 || ww > 127) continue;
                    acc += wp[(dh + 1) * 3 + (dw + 1)] * xc[hh * 128 + ww];
                }
            }
            cat_s[wl * LDSK + c] = f2bf(acc);
        }
    }
    // ---- build cat channels 256..511: quadrant-constant splat -------------
    {
        const int qh = h >> 6, qw = wt;
        const unsigned short bv =
            f2bf(coarse[((size_t)b * 256 + tid) * 4 + qh * 2 + qw]);
        for (int n = 0; n < 64; ++n)
            cat_s[n * LDSK + 256 + tid] = bv;
    }
    __syncthreads();

    // ---- WMMA GEMM: D[m,n] += wa[m,k] * cat[n][k] -------------------------
    const int lane = tid & 31;
    const int wv   = tid >> 5;          // 0..7
    const int m0   = wv * 64;
    const int lm   = lane & 15;
    const int lh   = lane >> 4;         // lane half selects K sub-chunks

    v8f acc[4][4] = {};
    for (int kt = 0; kt < 16; ++kt) {
        const int k0 = kt * 32;
        v16bf A[4], Bf[4];
        #pragma unroll
        for (int mt = 0; mt < 4; ++mt) {
            // A 16x32 bf16 layout: lanes 0-15 hold K {0..7,16..23}, 16-31 {8..15,24..31}
            const unsigned short* pa =
                wa_bf + (size_t)(m0 + mt * 16 + lm) * 512 + k0 + lh * 8;
            A[mt] = ldpair(pa, pa + 16);
        }
        #pragma unroll
        for (int nt = 0; nt < 4; ++nt) {
            // B 32x16 bf16 layout: lanes 0-15 hold K {0..7,16..23}, 16-31 {8..15,24..31}
            const unsigned short* pb =
                cat_s + (nt * 16 + lm) * LDSK + k0 + lh * 8;
            Bf[nt] = ldpair(pb, pb + 16);
        }
        if (kt + 1 < 16)   // pull next wa K-slice toward the WGP
            __builtin_prefetch(wa_bf + (size_t)(m0 + lm) * 512 + k0 + 32, 0, 1);
        #pragma unroll
        for (int mt = 0; mt < 4; ++mt)
            #pragma unroll
            for (int nt = 0; nt < 4; ++nt)
                acc[mt][nt] = __builtin_amdgcn_wmma_f32_16x16x32_bf16(
                    false, A[mt], false, Bf[nt], (short)0, acc[mt][nt],
                    false, false);
    }

    // ---- epilogue: bias + exact GELU + modulate by x ----------------------
    const size_t ximg = (size_t)b * 512 * HW + (size_t)h * 128 + w0;
    #pragma unroll
    for (int mt = 0; mt < 4; ++mt) {
        #pragma unroll
        for (int nt = 0; nt < 4; ++nt) {
            const int n = nt * 16 + lm;
            #pragma unroll
            for (int j = 0; j < 8; ++j) {
                const int o = m0 + mt * 16 + j + (lh ? 8 : 0);
                float v = acc[mt][nt][j] + ba[o];
                float g = 0.5f * v * (1.0f + erff(v * 0.70710678118654752f));
                const size_t idx = ximg + (size_t)o * HW + n;
                out[idx] = g * x[idx];
            }
        }
    }
}

// ---------------------------------------------------------------------------
extern "C" void kernel_launch(void* const* d_in, const int* in_sizes, int n_in,
                              void* d_out, int out_size, void* d_ws, size_t ws_size,
                              hipStream_t stream) {
    const float* x  = (const float*)d_in[0];
    const float* w1 = (const float*)d_in[1];
    const float* b1 = (const float*)d_in[2];
    const float* w2 = (const float*)d_in[3];
    const float* b2 = (const float*)d_in[4];
    const float* w3 = (const float*)d_in[5];
    const float* b3 = (const float*)d_in[6];
    const float* w4 = (const float*)d_in[7];
    const float* b4 = (const float*)d_in[8];
    const float* wa = (const float*)d_in[9];
    const float* ba = (const float*)d_in[10];

    unsigned short* wa_bf = (unsigned short*)d_ws;                 // 512 KB
    float* p2     = (float*)((char*)d_ws + 512 * 512 * 2);         // 16 KB
    float* coarse = p2 + 4 * 256 * 4;                              // 16 KB

    safm_cvt_wa  <<<1024, 256, 0, stream>>>(wa, wa_bf);
    safm_pool_max<<<4096, 256, 0, stream>>>(x, p2);
    safm_coarse  <<<1,    256, 0, stream>>>(p2, w2, b2, w3, b3, w4, b4, coarse);
    safm_main    <<<1024, 256, 0, stream>>>(x, w1, b1, coarse, wa_bf, ba,
                                            (float*)d_out);
}
